// BahdanauAttention_29472065585933
// MI455X (gfx1250) — matrix operand order repaired and checked by reading, not compiled, not run on hardware
//
#include <hip/hip_runtime.h>
#include <math.h>

#define BB   1024
#define LL   100
#define CC   1024
#define UU   512
#define GG   10
#define WINN 3
#define NLOC 9     // WIN*WIN
#define NPAD 128   // Wa output columns padded 100 -> 128 (multiple of 128)

typedef __attribute__((ext_vector_type(16))) _Float16 v16h;
typedef __attribute__((ext_vector_type(8)))  _Float16 v8h;
typedef __attribute__((ext_vector_type(8)))  float    v8f;

// ---------------------------------------------------------------------------
// WMMA GEMM:  C[M,N] = act( A[M,K] * W[K,N] + bias[N] + extra[m/div, N] )
//   A  : f16 row-major (M x K), K-contiguous
//   WT : f16 weights pre-transposed (N x K), K-contiguous
// Register-blocked 2x2: each wave computes a 32(M) x 32(N) tile with 4
// accumulators; A/B fragments are each reused twice -> 8 b128 loads per
// 4 wmma.  8 waves/block arranged 2(M) x 4(N) -> 64 x 128 per block.
// Requires M%64==0, N%128==0, K%64==0 -> no per-lane guards, EXEC all-ones.
// Fragment layouts as loaded here (contiguous k order, not the ISA table order):
//   A 16x32 f16 : lane half h holds K = h*16+{0..15}, row = lane&15
//   B 32x16 f16 : lane half h holds K = h*16+{0..15}, col = lane&15
//   D 16x16 f32 : VGPR i -> row = half*8 + i, col = lane&15
// ---------------------------------------------------------------------------
__global__ __launch_bounds__(256) void gemm_wmma_f16(
    const _Float16* __restrict__ A,   // M x K
    const _Float16* __restrict__ WT,  // N x K (transposed)
    const float* __restrict__ bias, const float* __restrict__ extra,
    int extra_div, float* __restrict__ C,
    int M, int N, int K, int do_tanh)
{
    const int lane = threadIdx.x & 31;
    const int wave = threadIdx.x >> 5;
    const int m16  = lane & 15;
    const int half = lane >> 4;

    const int m0 = blockIdx.y * 64  + (wave >> 2) * 32;   // wave M origin
    const int n0 = blockIdx.x * 128 + (wave & 3) * 32;    // wave N origin

    const _Float16* __restrict__ Arow0 = A  + (size_t)(m0 + m16) * K;
    const _Float16* __restrict__ Arow1 = A  + (size_t)(m0 + 16 + m16) * K;
    const _Float16* __restrict__ Brow0 = WT + (size_t)(n0 + m16) * K;
    const _Float16* __restrict__ Brow1 = WT + (size_t)(n0 + 16 + m16) * K;

    v8f acc00 = {}, acc01 = {}, acc10 = {}, acc11 = {};
#pragma unroll 2
    for (int k0 = 0; k0 < K; k0 += 32) {
        __builtin_prefetch(Brow0 + k0 + 256, 0, 1);  // global_prefetch ahead
        const v8h a0lo = *(const v8h*)(Arow0 + k0 + half * 16);
        const v8h a0hi = *(const v8h*)(Arow0 + k0 + half * 16 + 8);
        const v8h a1lo = *(const v8h*)(Arow1 + k0 + half * 16);
        const v8h a1hi = *(const v8h*)(Arow1 + k0 + half * 16 + 8);
        const v16h a0 = __builtin_shufflevector(
            a0lo, a0hi, 0, 1, 2, 3, 4, 5, 6, 7, 8, 9, 10, 11, 12, 13, 14, 15);
        const v16h a1 = __builtin_shufflevector(
            a1lo, a1hi, 0, 1, 2, 3, 4, 5, 6, 7, 8, 9, 10, 11, 12, 13, 14, 15);
        const v16h b0 = *(const v16h*)(Brow0 + k0 + half * 16);
        const v16h b1 = *(const v16h*)(Brow1 + k0 + half * 16);
        acc00 = __builtin_amdgcn_wmma_f32_16x16x32_f16(
                    false, a0, false, b0, (short)0, acc00, false, false);
        acc01 = __builtin_amdgcn_wmma_f32_16x16x32_f16(
                    false, a0, false, b1, (short)0, acc01, false, false);
        acc10 = __builtin_amdgcn_wmma_f32_16x16x32_f16(
                    false, a1, false, b0, (short)0, acc10, false, false);
        acc11 = __builtin_amdgcn_wmma_f32_16x16x32_f16(
                    false, a1, false, b1, (short)0, acc11, false, false);
    }

    const int col0 = n0 + m16;
    const int col1 = n0 + 16 + m16;
    const float bv0 = bias ? bias[col0] : 0.0f;
    const float bv1 = bias ? bias[col1] : 0.0f;
#pragma unroll
    for (int i = 0; i < 8; ++i) {
        const int row0 = m0 + half * 8 + i;
        const int row1 = row0 + 16;
        float v00 = acc00[i] + bv0;
        float v01 = acc01[i] + bv1;
        float v10 = acc10[i] + bv0;
        float v11 = acc11[i] + bv1;
        if (extra) {
            const float* e0 = extra + (size_t)(row0 / extra_div) * N;
            const float* e1 = extra + (size_t)(row1 / extra_div) * N;
            v00 += e0[col0]; v01 += e0[col1];
            v10 += e1[col0]; v11 += e1[col1];
        }
        if (do_tanh) {
            v00 = tanhf(v00); v01 = tanhf(v01);
            v10 = tanhf(v10); v11 = tanhf(v11);
        }
        C[(size_t)row0 * N + col0] = v00;
        C[(size_t)row0 * N + col1] = v01;
        C[(size_t)row1 * N + col0] = v10;
        C[(size_t)row1 * N + col1] = v11;
    }
}

// ---------------------------------------------------------------------------
// Transpose + convert weights:  in (K x N, f32)  ->  out (Nout x K, f16),
// zero-padded for n >= N.
// ---------------------------------------------------------------------------
__global__ __launch_bounds__(256) void transpose_to_f16(
    const float* __restrict__ in, _Float16* __restrict__ out,
    int K, int N, int Nout)
{
    const size_t total = (size_t)Nout * K;
    for (size_t idx = (size_t)blockIdx.x * blockDim.x + threadIdx.x;
         idx < total; idx += (size_t)gridDim.x * blockDim.x) {
        const int n = (int)(idx / K);
        const int k = (int)(idx - (size_t)n * K);
        out[idx] = (n < N) ? (_Float16)in[(size_t)k * N + n] : (_Float16)0.0f;
    }
}

// f32 -> f16 elementwise copy
__global__ __launch_bounds__(256) void cvt_f16_kernel(
    const float* __restrict__ in, _Float16* __restrict__ out, size_t n)
{
    for (size_t i = (size_t)blockIdx.x * blockDim.x + threadIdx.x;
         i < n; i += (size_t)gridDim.x * blockDim.x)
        out[i] = (_Float16)in[i];
}

// ---------------------------------------------------------------------------
// p_t = sigmoid(tanhWa @ Wb) * 8 + 1 ; starts = clamp(int(p_t - 1), 0, 7)
// tanhWa is padded to stride NPAD.
// ---------------------------------------------------------------------------
__global__ void pt_kernel(const float* __restrict__ tanhWa,
                          const float* __restrict__ Wb,
                          int* __restrict__ starts)
{
    const int b = blockIdx.x * blockDim.x + threadIdx.x;
    if (b >= BB) return;
    float a0 = 0.f, a1 = 0.f;
    const float* t = tanhWa + (size_t)b * NPAD;
    for (int j = 0; j < LL; ++j) {
        const float v = t[j];
        a0 += v * Wb[j * 2 + 0];
        a1 += v * Wb[j * 2 + 1];
    }
    float p0 = 1.f / (1.f + expf(-a0)); p0 = p0 * 8.f + 1.f;
    float p1 = 1.f / (1.f + expf(-a1)); p1 = p1 * 8.f + 1.f;
    int s0 = (int)(p0 - 1.f);
    int s1 = (int)(p1 - 1.f);
    s0 = s0 < 0 ? 0 : (s0 > 7 ? 7 : s0);
    s1 = s1 < 0 ? 0 : (s1 > 7 ? 7 : s1);
    starts[b * 2 + 0] = s0;
    starts[b * 2 + 1] = s1;
}

// ---------------------------------------------------------------------------
// Gather 3x3xC window; emit f32 (for ctx) and f16 (GEMM A operand) copies.
// ---------------------------------------------------------------------------
__global__ __launch_bounds__(256) void gather_kernel(
    const float* __restrict__ features, const int* __restrict__ starts,
    float* __restrict__ local, _Float16* __restrict__ local_h)
{
    const int b  = blockIdx.x;
    const int s0 = starts[b * 2];
    const int s1 = starts[b * 2 + 1];
    for (int idx = threadIdx.x; idx < NLOC * CC; idx += blockDim.x) {
        const int j  = idx / CC;
        const int c  = idx - j * CC;
        const int jj = j / WINN;
        const int kk = j - jj * WINN;
        const int row = (s0 + jj) * GG + (s1 + kk);
        const float v = features[((size_t)b * LL + row) * CC + c];
        const size_t di = ((size_t)b * NLOC + j) * CC + c;
        local[di]   = v;
        local_h[di] = (_Float16)v;
    }
}

// ---------------------------------------------------------------------------
// Per batch: logits = score @ V1 + b ; softmax over 9 ; * gaussian ;
// ctx = sum_j attn_j * local_j ; tcat = tanh([ctx, hidden]) (f16) ; emit attn.
// ---------------------------------------------------------------------------
__global__ __launch_bounds__(256) void attn_ctx_kernel(
    const float* __restrict__ score, const float* __restrict__ local,
    const float* __restrict__ V1_w, const float* __restrict__ V1_b,
    const float* __restrict__ hidden,
    float* __restrict__ attn_out, _Float16* __restrict__ tcat_h)
{
    __shared__ float red[NLOC * 256];
    __shared__ float attn_s[NLOC];
    const int b = blockIdx.x;
    const int t = threadIdx.x;

    float part[NLOC];
#pragma unroll
    for (int j = 0; j < NLOC; ++j) part[j] = 0.f;
    for (int u = t; u < UU; u += 256) {
        const float v1 = V1_w[u];
#pragma unroll
        for (int j = 0; j < NLOC; ++j)
            part[j] += score[((size_t)b * NLOC + j) * UU + u] * v1;
    }
#pragma unroll
    for (int j = 0; j < NLOC; ++j) red[j * 256 + t] = part[j];
    __syncthreads();
    for (int s = 128; s > 0; s >>= 1) {
        if (t < s) {
#pragma unroll
            for (int j = 0; j < NLOC; ++j)
                red[j * 256 + t] += red[j * 256 + t + s];
        }
        __syncthreads();
    }
    if (t == 0) {
        float logits[NLOC], mx = -1e30f;
        for (int j = 0; j < NLOC; ++j) {
            logits[j] = red[j * 256] + V1_b[0];
            mx = fmaxf(mx, logits[j]);
        }
        float sum = 0.f;
        for (int j = 0; j < NLOC; ++j) {
            logits[j] = expf(logits[j] - mx);
            sum += logits[j];
        }
        for (int j = 0; j < NLOC; ++j) {
            const int jj = j / WINN, kk = j - jj * WINN;
            const float d2 = (jj - 1.5f) * (jj - 1.5f) + (kk - 1.5f) * (kk - 1.5f);
            const float g  = expf(-2.f * d2);            // exp(-d2 / (0.5*D*D))
            const float a  = (logits[j] / sum) * g;
            attn_s[j] = a;
            attn_out[(size_t)b * NLOC + j] = a;
        }
    }
    __syncthreads();
    for (int c = t; c < CC; c += 256) {
        float s = 0.f;
#pragma unroll
        for (int j = 0; j < NLOC; ++j)
            s += attn_s[j] * local[((size_t)b * NLOC + j) * CC + c];
        tcat_h[(size_t)b * (CC + UU) + c] = (_Float16)tanhf(s);
    }
    for (int u = t; u < UU; u += 256)
        tcat_h[(size_t)b * (CC + UU) + CC + u] =
            (_Float16)tanhf(hidden[(size_t)b * UU + u]);
}

// ---------------------------------------------------------------------------
extern "C" void kernel_launch(void* const* d_in, const int* in_sizes, int n_in,
                              void* d_out, int out_size, void* d_ws, size_t ws_size,
                              hipStream_t stream)
{
    const float* features = (const float*)d_in[0];   // (B,L,C)
    const float* hidden   = (const float*)d_in[1];   // (B,U)
    const float* W1_w     = (const float*)d_in[2];   // (C,U)
    const float* W1_b     = (const float*)d_in[3];   // (U,)
    const float* W2_w     = (const float*)d_in[4];   // (U,U)
    const float* W2_b     = (const float*)d_in[5];   // (U,)
    const float* V1_w     = (const float*)d_in[6];   // (U,1)
    const float* V1_b     = (const float*)d_in[7];   // (1,)
    const float* W3_w     = (const float*)d_in[8];   // (C+U,U)
    const float* W3_b     = (const float*)d_in[9];   // (U,)
    const float* Wa       = (const float*)d_in[10];  // (U,100)
    const float* Wb       = (const float*)d_in[11];  // (100,2)

    float* out      = (float*)d_out;                 // (B,U)
    float* attn_out = out + (size_t)BB * UU;         // (B,9)

    // ---- workspace layout (256B-aligned sub-buffers) ----
    char*  base = (char*)d_ws;
    size_t off  = 0;
    auto alloc = [&](size_t bytes) -> char* {
        char* p = base + off;
        off = (off + bytes + 255) & ~(size_t)255;
        return p;
    };
    _Float16* hidden_h = (_Float16*)alloc((size_t)BB * UU * 2);
    _Float16* WahT     = (_Float16*)alloc((size_t)NPAD * UU * 2);
    _Float16* W1T      = (_Float16*)alloc((size_t)UU * CC * 2);
    _Float16* W2T      = (_Float16*)alloc((size_t)UU * UU * 2);
    _Float16* W3T      = (_Float16*)alloc((size_t)UU * (CC + UU) * 2);
    float*    tanhWa   = (float*)alloc((size_t)BB * NPAD * 4);
    float*    pre2     = (float*)alloc((size_t)BB * UU * 4);
    float*    local    = (float*)alloc((size_t)BB * NLOC * CC * 4);
    _Float16* local_h  = (_Float16*)alloc((size_t)BB * NLOC * CC * 2);
    float*    score    = (float*)alloc((size_t)BB * NLOC * UU * 4);
    _Float16* tcat_h   = (_Float16*)alloc((size_t)BB * (CC + UU) * 2);
    int*      starts   = (int*)alloc((size_t)BB * 2 * 4);

    const dim3 blk(256);

    // ---- operand preparation (f16 conversion / weight transposes) ----
    cvt_f16_kernel<<<dim3(512), blk, 0, stream>>>(hidden, hidden_h, (size_t)BB * UU);
    transpose_to_f16<<<dim3(64),  blk, 0, stream>>>(Wa,   WahT, UU,      LL, NPAD);
    transpose_to_f16<<<dim3(256), blk, 0, stream>>>(W1_w, W1T,  CC,      UU, UU);
    transpose_to_f16<<<dim3(256), blk, 0, stream>>>(W2_w, W2T,  UU,      UU, UU);
    transpose_to_f16<<<dim3(256), blk, 0, stream>>>(W3_w, W3T,  CC + UU, UU, UU);

    // 1) tanhWa = tanh(hidden @ Wa)              [1024 x 128, K=512]
    gemm_wmma_f16<<<dim3(NPAD / 128, BB / 64), blk, 0, stream>>>(
        hidden_h, WahT, nullptr, nullptr, 1, tanhWa, BB, NPAD, UU, 1);

    // 2) p_t -> window starts
    pt_kernel<<<dim3(BB / 256), blk, 0, stream>>>(tanhWa, Wb, starts);

    // 3) gather local windows (f32 + f16)
    gather_kernel<<<dim3(BB), blk, 0, stream>>>(features, starts, local, local_h);

    // 4) pre2 = hidden @ W2 + b2                 [1024 x 512, K=512]
    gemm_wmma_f16<<<dim3(UU / 128, BB / 64), blk, 0, stream>>>(
        hidden_h, W2T, W2_b, nullptr, 1, pre2, BB, UU, UU, 0);

    // 5) score = tanh(local @ W1 + b1 + pre2[b]) [9216 x 512, K=1024]
    gemm_wmma_f16<<<dim3(UU / 128, (BB * NLOC) / 64), blk, 0, stream>>>(
        local_h, W1T, W1_b, pre2, NLOC, score, BB * NLOC, UU, CC, 1);

    // 6) logits / softmax*gauss / ctx / tcat(f16) ; emit attn
    attn_ctx_kernel<<<dim3(BB), blk, 0, stream>>>(
        score, local, V1_w, V1_b, hidden, attn_out, tcat_h);

    // 7) out = tcat @ W3 + b3                    [1024 x 512, K=1536]
    gemm_wmma_f16<<<dim3(UU / 128, BB / 64), blk, 0, stream>>>(
        tcat_h, W3T, W3_b, nullptr, 1, out, BB, UU, CC + UU, 0);
}
